// TransformerDo_36490042147473
// MI455X (gfx1250) — matrix operand order repaired and checked by reading, not compiled, not run on hardware
//
#include <hip/hip_runtime.h>
#include <hip/hip_bf16.h>
#include <math.h>
#include <stdint.h>

// ---------------------------------------------------------------------------
// GPT-2-small-ish forward (B=1, L=2048, D=768, H=12, Dh=64, N=4, F=3072,
// V=50257) for MI455X (gfx1250, wave32).
//  - All GEMMs: V_WMMA_F32_16X16X32_BF16, fp32 accumulate.
//  - Operands pre-converted to bf16 (weights transposed to [N,K] once/launch).
//  - Tile staging: GLOBAL_LOAD_ASYNC_TO_LDS_B128 (ASYNCcnt) w/ double-buffered
//    LDS; residual stream and softmax statistics stay fp32.
// ---------------------------------------------------------------------------

#define LSEQ 2048
#define DMODEL 768
#define NHEAD 12
#define DHEAD 64
#define NLAYER 4
#define FFDIM 3072
#define VOCAB 50257

#define TM 128      // block tile M
#define TN 128      // block tile N
#define TK 32       // K step (bf16 WMMA K)
#define KSTRIDE 40  // padded LDS row stride (elements): 80B rows, 16B aligned

typedef __bf16 bf16_t;
typedef __bf16 v16bf __attribute__((ext_vector_type(16)));
typedef __bf16 v8bf  __attribute__((ext_vector_type(8)));
typedef float  v8f   __attribute__((ext_vector_type(8)));

union FragBF { v16bf v; v8bf h[2]; };

__device__ inline float gelu_tanh(float x) {
  return 0.5f * x * (1.0f + tanhf(0.7978845608028654f * (x + 0.044715f * x * x * x)));
}

// Low 32 bits of a generic pointer to LDS = byte offset in the LDS window.
__device__ inline uint32_t lds_addr(const void* p) {
  return (uint32_t)(uintptr_t)p;
}

// Async 16B copy global -> LDS (per-lane), tracked by ASYNCcnt.
__device__ inline void async_copy_b128(uint32_t lds_off, const void* gaddr) {
  asm volatile("global_load_async_to_lds_b128 %0, %1, off"
               :: "v"(lds_off), "v"(gaddr) : "memory");
}

// ---------------------------------------------------------------------------
// WMMA GEMM: acc = A[M,K](bf16,row-major,lda) x B[K,N] where B is stored
// transposed: B(k,n) = Bt[n*ldb + k]  (bf16, [N,K] row-major).
// Epilogue: y = alpha*acc; optional GELU; then any of:
//   Cf (fp32, row-major, + optional fp32 residual)
//   Cb (bf16, row-major)            [transCb==0]
//   Cb (bf16, transposed [N,M])     [transCb==1]
// Block: 256 threads / 8 waves; tile 128x128; wave tile 32x64 (2x4 WMMA).
// K must be a multiple of 32; M a multiple of 128; N guarded.
// ---------------------------------------------------------------------------
__global__ __launch_bounds__(256) void gemm_wmma_async(
    const bf16_t* __restrict__ A, int lda,
    const bf16_t* __restrict__ Bt, int ldb,
    const float* __restrict__ resid, int ldr,
    float* __restrict__ Cf, int ldcf,
    bf16_t* __restrict__ Cb, int ldcb, int transCb,
    int M, int Nn, int K, float alpha, int gelu)
{
  __shared__ bf16_t As[2][TM * KSTRIDE];
  __shared__ bf16_t Bs[2][TN * KSTRIDE];

  const int tid  = threadIdx.x;
  const int lane = tid & 31;
  const int wave = tid >> 5;
  const int wm   = wave & 3;   // 0..3 over M (32 rows each)
  const int wn   = wave >> 2;  // 0..1 over N (64 cols each)
  const int tileM = blockIdx.y * TM;
  const int tileN = blockIdx.x * TN;

  // Each thread copies 2 A-chunks + 2 B-chunks (16B each) per stage.
  // Chunk c (0..511): row = c>>2, k-offset = (c&3)*8.
  const int rowA0 = tid >> 2;            // 0..63
  const int rowA1 = rowA0 + 64;          // 64..127
  const int kc    = (tid & 3) * 8;       // 0,8,16,24
  const uint32_t asBase = lds_addr(&As[0][0]);
  const uint32_t bsBase = lds_addr(&Bs[0][0]);
  const uint32_t bufA   = (uint32_t)(TM * KSTRIDE * 2);  // bytes per A buffer
  const uint32_t bufB   = (uint32_t)(TN * KSTRIDE * 2);

  int nB0 = tileN + rowA0; if (nB0 >= Nn) nB0 = Nn - 1;  // clamp keeps counts uniform
  int nB1 = tileN + rowA1; if (nB1 >= Nn) nB1 = Nn - 1;

  auto issue_stage = [&](int kb, int buf) {
    const bf16_t* gA0 = A  + (size_t)(tileM + rowA0) * lda + kb + kc;
    const bf16_t* gA1 = A  + (size_t)(tileM + rowA1) * lda + kb + kc;
    const bf16_t* gB0 = Bt + (size_t)nB0 * ldb + kb + kc;
    const bf16_t* gB1 = Bt + (size_t)nB1 * ldb + kb + kc;
    const uint32_t aoff = (uint32_t)buf * bufA;
    const uint32_t boff = (uint32_t)buf * bufB;
    async_copy_b128(asBase + aoff + (uint32_t)(rowA0 * KSTRIDE + kc) * 2, gA0);
    async_copy_b128(asBase + aoff + (uint32_t)(rowA1 * KSTRIDE + kc) * 2, gA1);
    async_copy_b128(bsBase + boff + (uint32_t)(rowA0 * KSTRIDE + kc) * 2, gB0);
    async_copy_b128(bsBase + boff + (uint32_t)(rowA1 * KSTRIDE + kc) * 2, gB1);
  };

  v8f acc[2][4];
#pragma unroll
  for (int i = 0; i < 2; ++i)
#pragma unroll
    for (int j = 0; j < 4; ++j)
#pragma unroll
      for (int r = 0; r < 8; ++r) acc[i][j][r] = 0.0f;

  const int nk = K / TK;
  issue_stage(0, 0);

  const int lmod = lane & 15;
  const int lhi  = lane >> 4;   // half-wave select
  const int kA   = lhi << 3;    // A frag: base K 0 / 8 (pairs; +16 for 2nd run)
  const int kB   = lhi << 3;    // B frag: base K 0 / 8 (pairs; +16 for 2nd run)

  for (int s = 0; s < nk; ++s) {
    const int cur = s & 1;
    if (s + 1 < nk) {
      issue_stage((s + 1) * TK, cur ^ 1);
      asm volatile("s_wait_asynccnt 0x4" ::: "memory");  // stage s complete
    } else {
      asm volatile("s_wait_asynccnt 0x0" ::: "memory");
    }
    __syncthreads();

    const bf16_t* as = &As[cur][0];
    const bf16_t* bs = &Bs[cur][0];

    FragBF bfr[4];
#pragma unroll
    for (int j = 0; j < 4; ++j) {
      const int brow = (wn * 64 + j * 16 + lmod) * KSTRIDE;
      bfr[j].h[0] = *(const v8bf*)&bs[brow + kB];
      bfr[j].h[1] = *(const v8bf*)&bs[brow + kB + 16];
    }
#pragma unroll
    for (int i = 0; i < 2; ++i) {
      const int arow = (wm * 32 + i * 16 + lmod) * KSTRIDE;
      FragBF a;
      a.h[0] = *(const v8bf*)&as[arow + kA];
      a.h[1] = *(const v8bf*)&as[arow + kA + 16];
#pragma unroll
      for (int j = 0; j < 4; ++j) {
        acc[i][j] = __builtin_amdgcn_wmma_f32_16x16x32_bf16(
            false, a.v, false, bfr[j].v, (short)0, acc[i][j], false, false);
      }
    }
    __syncthreads();
  }

  // ---- epilogue ---------------------------------------------------------
  const int rofs = lhi << 3;  // +8 rows for lanes 16-31
#pragma unroll
  for (int i = 0; i < 2; ++i) {
#pragma unroll
    for (int j = 0; j < 4; ++j) {
      const int col  = tileN + wn * 64 + j * 16 + lmod;
      const int row0 = tileM + wm * 32 + i * 16 + rofs;
      if (col >= Nn) continue;
      float y[8];
#pragma unroll
      for (int r = 0; r < 8; ++r) {
        float t = acc[i][j][r] * alpha;
        if (gelu) t = gelu_tanh(t);
        y[r] = t;
      }
      if (Cf) {
#pragma unroll
        for (int r = 0; r < 8; ++r) {
          float t = y[r];
          if (resid) t += resid[(size_t)(row0 + r) * ldr + col];
          Cf[(size_t)(row0 + r) * ldcf + col] = t;
        }
      }
      if (Cb) {
        if (transCb) {  // [N,M]: 8 consecutive rows -> one 16B store
          v8bf pk;
#pragma unroll
          for (int r = 0; r < 8; ++r) pk[r] = (__bf16)y[r];
          *(v8bf*)(Cb + (size_t)col * ldcb + row0) = pk;
        } else {
#pragma unroll
          for (int r = 0; r < 8; ++r)
            Cb[(size_t)(row0 + r) * ldcb + col] = (__bf16)y[r];
        }
      }
    }
  }
}

// ---------------------------------------------------------------------------
// fp32 -> bf16 convert (no transpose): embed table for the logits GEMM.
// ---------------------------------------------------------------------------
__global__ __launch_bounds__(256) void convert_kernel(
    const float* __restrict__ X, bf16_t* __restrict__ Y, int n)
{
  int idx = blockIdx.x * 256 + threadIdx.x;
  if (idx < n) Y[idx] = (__bf16)X[idx];
}

// ---------------------------------------------------------------------------
// fp32 [K,N] -> bf16 [N,K] (transpose). Coalesced reads, scattered b16 writes
// (L2 absorbs; runs once per launch over 28M weight elements).
// ---------------------------------------------------------------------------
__global__ __launch_bounds__(256) void convertT_kernel(
    const float* __restrict__ W, bf16_t* __restrict__ Wt, int K, int N)
{
  int idx = blockIdx.x * 256 + threadIdx.x;
  if (idx < K * N) {
    int k = idx / N;
    int n = idx - k * N;
    Wt[(size_t)n * K + k] = (__bf16)W[idx];
  }
}

// ---------------------------------------------------------------------------
// Embedding gather: x[l,d] = embed[tokens[l], d] + pos[l, d]  (fp32)
// ---------------------------------------------------------------------------
__global__ __launch_bounds__(256) void embed_kernel(
    const int* __restrict__ tokens, const float* __restrict__ embed,
    const float* __restrict__ pos, float* __restrict__ x)
{
  int idx = blockIdx.x * 256 + threadIdx.x;
  if (idx < LSEQ * DMODEL) {
    int l = idx / DMODEL;
    int d = idx - l * DMODEL;
    x[idx] = embed[(size_t)tokens[l] * DMODEL + d] + pos[idx];
  }
}

// ---------------------------------------------------------------------------
// LayerNorm over last dim; fp32 in, bf16 out. One block (256 thr) per row.
// ---------------------------------------------------------------------------
__global__ __launch_bounds__(256) void layernorm_kernel(
    const float* __restrict__ x, const float* __restrict__ g,
    bf16_t* __restrict__ out, int Dd)
{
  __shared__ float red[256];
  const int row = blockIdx.x;
  const int tid = threadIdx.x;
  const float* xr = x + (size_t)row * Dd;

  float s = 0.0f;
  for (int i = tid; i < Dd; i += 256) s += xr[i];
  red[tid] = s; __syncthreads();
  for (int o = 128; o > 0; o >>= 1) { if (tid < o) red[tid] += red[tid + o]; __syncthreads(); }
  const float mu = red[0] / (float)Dd;
  __syncthreads();

  float vs = 0.0f;
  for (int i = tid; i < Dd; i += 256) { float dd = xr[i] - mu; vs += dd * dd; }
  red[tid] = vs; __syncthreads();
  for (int o = 128; o > 0; o >>= 1) { if (tid < o) red[tid] += red[tid + o]; __syncthreads(); }
  const float rinv = rsqrtf(red[0] / (float)Dd + 1e-6f);

  for (int i = tid; i < Dd; i += 256)
    out[(size_t)row * Dd + i] = (__bf16)((xr[i] - mu) * rinv * g[i]);
}

// ---------------------------------------------------------------------------
// Causal softmax: P[row, 0..row] = softmax(S[row, 0..row]) in bf16; rest 0.
// fp32 max/sum statistics; one block per row.
// ---------------------------------------------------------------------------
__global__ __launch_bounds__(256) void softmax_causal_kernel(
    const float* __restrict__ S, bf16_t* __restrict__ P, int Ln)
{
  __shared__ float red[256];
  const int row = blockIdx.x;
  const int tid = threadIdx.x;
  const float* sr = S + (size_t)row * Ln;
  bf16_t* pr = P + (size_t)row * Ln;
  const int limit = row;  // inclusive

  float mx = -3.402823466e+38f;
  for (int i = tid; i <= limit; i += 256) mx = fmaxf(mx, sr[i]);
  red[tid] = mx; __syncthreads();
  for (int o = 128; o > 0; o >>= 1) { if (tid < o) red[tid] = fmaxf(red[tid], red[tid + o]); __syncthreads(); }
  const float m = red[0];
  __syncthreads();

  float sum = 0.0f;
  for (int i = tid; i <= limit; i += 256) sum += __expf(sr[i] - m);
  red[tid] = sum; __syncthreads();
  for (int o = 128; o > 0; o >>= 1) { if (tid < o) red[tid] += red[tid + o]; __syncthreads(); }
  const float inv = 1.0f / red[0];

  for (int i = tid; i <= limit; i += 256) pr[i] = (__bf16)(__expf(sr[i] - m) * inv);
  for (int i = limit + 1 + tid; i < Ln; i += 256) pr[i] = (__bf16)0.0f;
}

// ---------------------------------------------------------------------------
extern "C" void kernel_launch(void* const* d_in, const int* in_sizes, int n_in,
                              void* d_out, int out_size, void* d_ws, size_t ws_size,
                              hipStream_t stream)
{
  (void)in_sizes; (void)n_in; (void)out_size; (void)ws_size;

  const int*   tokens = (const int*)  d_in[0];
  const float* embedW = (const float*)d_in[1];
  const float* posW   = (const float*)d_in[2];
  const float* WqA    = (const float*)d_in[3];   // [N,D,768]
  const float* WkA    = (const float*)d_in[4];
  const float* WvA    = (const float*)d_in[5];
  const float* WoA    = (const float*)d_in[6];   // [N,768,D]
  const float* ln1A   = (const float*)d_in[7];
  const float* W1A    = (const float*)d_in[8];   // [N,D,F]
  const float* W2A    = (const float*)d_in[9];   // [N,F,D]
  const float* ln2A   = (const float*)d_in[10];
  const float* outLn  = (const float*)d_in[11];
  float* logits = (float*)d_out;

  const size_t LD = (size_t)LSEQ * DMODEL;
  const size_t DD = (size_t)DMODEL * DMODEL;
  const size_t DF = (size_t)DMODEL * FFDIM;
  const size_t layerW = 4 * DD + 2 * DF;

  // ---- workspace carve-out ---------------------------------------------
  float* f = (float*)d_ws;
  float* x  = f; f += LD;                    // residual stream
  float* x2 = f; f += LD;                    // post-attention stream
  float* S  = f; f += (size_t)LSEQ * LSEQ;   // per-head scores (fp32)
  bf16_t* b = (bf16_t*)f;
  bf16_t* h  = b; b += LD;                   // LN output
  bf16_t* q  = b; b += LD;
  bf16_t* k  = b; b += LD;
  bf16_t* vT = b; b += LD;                   // V transposed: [768, L]
  bf16_t* o  = b; b += LD;                   // attention out (pre-Wo)
  bf16_t* P  = b; b += (size_t)LSEQ * LSEQ;  // softmax probs
  bf16_t* z  = b; b += (size_t)LSEQ * FFDIM; // FFN hidden
  bf16_t* embedB = b; b += (size_t)VOCAB * DMODEL;
  bf16_t* wT = b;                            // transposed bf16 weights, [N,K]

  auto gemm = [&](const bf16_t* A, int lda, const bf16_t* Bt, int ldb,
                  const float* resid, int ldr, float* Cf, int ldcf,
                  bf16_t* Cb, int ldcb, int transCb,
                  int M, int Nn, int K, float alpha, int gelu) {
    dim3 grid((Nn + TN - 1) / TN, M / TM);
    gemm_wmma_async<<<grid, 256, 0, stream>>>(A, lda, Bt, ldb, resid, ldr,
                                              Cf, ldcf, Cb, ldcb, transCb,
                                              M, Nn, K, alpha, gelu);
  };
  auto convT = [&](const float* W, bf16_t* Wt, int K, int N) {
    convertT_kernel<<<(K * N + 255) / 256, 256, 0, stream>>>(W, Wt, K, N);
  };

  // ---- one-time (per launch) weight preparation ------------------------
  convert_kernel<<<((int)(VOCAB * DMODEL) + 255) / 256, 256, 0, stream>>>(
      embedW, embedB, VOCAB * DMODEL);
  for (int l = 0; l < NLAYER; ++l) {
    bf16_t* wb = wT + (size_t)l * layerW;
    convT(WqA + l * DD, wb + 0 * DD, DMODEL, DMODEL);
    convT(WkA + l * DD, wb + 1 * DD, DMODEL, DMODEL);
    convT(WvA + l * DD, wb + 2 * DD, DMODEL, DMODEL);
    convT(WoA + l * DD, wb + 3 * DD, DMODEL, DMODEL);
    convT(W1A + l * DF, wb + 4 * DD, DMODEL, FFDIM);          // [F, D]
    convT(W2A + l * DF, wb + 4 * DD + DF, FFDIM, DMODEL);     // [D, F]
  }

  // x = embed[tokens] + pos
  embed_kernel<<<(LSEQ * DMODEL + 255) / 256, 256, 0, stream>>>(tokens, embedW, posW, x);

  const float attScale = 1.0f / sqrtf((float)DHEAD);

  for (int l = 0; l < NLAYER; ++l) {
    bf16_t* wqT = wT + (size_t)l * layerW;
    bf16_t* wkT = wqT + DD;
    bf16_t* wvT = wkT + DD;
    bf16_t* woT = wvT + DD;
    bf16_t* w1T = woT + DD;
    bf16_t* w2T = w1T + DF;
    const float* g1 = ln1A + (size_t)l * DMODEL;
    const float* g2 = ln2A + (size_t)l * DMODEL;

    // h = LN(x, g1)
    layernorm_kernel<<<LSEQ, 256, 0, stream>>>(x, g1, h, DMODEL);

    // q,k = h x W (bf16 out);  v = h x Wv written transposed [768, L]
    gemm(h, DMODEL, wqT, DMODEL, nullptr, 0, nullptr, 0, q, DMODEL, 0,
         LSEQ, DMODEL, DMODEL, 1.0f, 0);
    gemm(h, DMODEL, wkT, DMODEL, nullptr, 0, nullptr, 0, k, DMODEL, 0,
         LSEQ, DMODEL, DMODEL, 1.0f, 0);
    gemm(h, DMODEL, wvT, DMODEL, nullptr, 0, nullptr, 0, vT, LSEQ, 1,
         LSEQ, DMODEL, DMODEL, 1.0f, 0);

    // attention, head by head (S/P buffers reused)
    for (int hd = 0; hd < NHEAD; ++hd) {
      // S = (q/sqrt(Dh)) x k^T : A = q_h [L,64], Bt = k_h [L,64] -> [L,L] fp32
      gemm(q + hd * DHEAD, DMODEL, k + hd * DHEAD, DMODEL,
           nullptr, 0, S, LSEQ, nullptr, 0, 0,
           LSEQ, LSEQ, DHEAD, attScale, 0);
      softmax_causal_kernel<<<LSEQ, 256, 0, stream>>>(S, P, LSEQ);
      // o_h = P x v_h : Bt = vT_h [64, L] -> bf16 [L,64] slice of o
      gemm(P, LSEQ, vT + (size_t)hd * DHEAD * LSEQ, LSEQ,
           nullptr, 0, nullptr, 0, o + hd * DHEAD, DMODEL, 0,
           LSEQ, DHEAD, LSEQ, 1.0f, 0);
    }

    // x2 = x + o x Wo   (fp32)
    gemm(o, DMODEL, woT, DMODEL, x, DMODEL, x2, DMODEL, nullptr, 0, 0,
         LSEQ, DMODEL, DMODEL, 1.0f, 0);

    // h = LN(x2, g2);  z = gelu(h x W1) bf16;  x = x2 + z x W2 fp32
    layernorm_kernel<<<LSEQ, 256, 0, stream>>>(x2, g2, h, DMODEL);
    gemm(h, DMODEL, w1T, DMODEL, nullptr, 0, nullptr, 0, z, FFDIM, 0,
         LSEQ, FFDIM, DMODEL, 1.0f, 1);
    gemm(z, FFDIM, w2T, FFDIM, x2, DMODEL, x, DMODEL, nullptr, 0, 0,
         LSEQ, DMODEL, FFDIM, 1.0f, 0);
  }

  // logits = LN(x, out_ln) x embed^T   [L,V] fp32
  layernorm_kernel<<<LSEQ, 256, 0, stream>>>(x, outLn, h, DMODEL);
  gemm(h, DMODEL, embedB, DMODEL, nullptr, 0, logits, VOCAB, nullptr, 0, 0,
       LSEQ, VOCAB, DMODEL, 1.0f, 0);
}
